// RowLSTM_23733989277863
// MI455X (gfx1250) — matrix operand order repaired and checked by reading, not compiled, not run on hardware
//
#include <hip/hip_runtime.h>
#include <math.h>

typedef _Float16 h16;
typedef __attribute__((ext_vector_type(4)))  _Float16 half4;
typedef __attribute__((ext_vector_type(8)))  _Float16 half8;
typedef __attribute__((ext_vector_type(16))) _Float16 half16;
typedef __attribute__((ext_vector_type(8)))  float    float8;

typedef __attribute__((ext_vector_type(4))) int i32x4;
typedef __attribute__((address_space(1))) i32x4 i32x4_g;  // global
typedef __attribute__((address_space(3))) i32x4 i32x4_l;  // LDS

constexpr int kW   = 122;   // width
constexpr int kH   = 122;   // height (scan length)
constexpr int kB   = 32;    // batch
constexpr int kHid = 64;    // hidden channels
constexpr int RS   = 72;    // LDS row stride (h16 units) -> conflict-free, rows 16B-aligned
constexpr int NR   = 130;   // rows: n' in [-1,128] stored at n'+1

#if defined(__HIP_DEVICE_COMPILE__) && \
    __has_builtin(__builtin_amdgcn_global_load_async_to_lds_b128) && \
    __has_builtin(__builtin_amdgcn_s_wait_asynccnt)
#define HAVE_ASYNC 1
#else
#define HAVE_ASYNC 0
#endif

__device__ __forceinline__ void cp16_g2l(h16* l, const h16* g) {
#if HAVE_ASYNC
  __builtin_amdgcn_global_load_async_to_lds_b128(
      (i32x4_g*)(i32x4*)const_cast<h16*>(g),
      (i32x4_l*)(i32x4*)l,
      0, 0);
#else
  *(half8*)l = *(const half8*)g;
#endif
}

// branch-free activations (v_exp_f32 + v_rcp_f32; no OCML exec-mask branches)
__device__ __forceinline__ float rcpf(float x)  { return __builtin_amdgcn_rcpf(x); }
__device__ __forceinline__ float sigf(float x)  { return rcpf(1.0f + __expf(-x)); }
__device__ __forceinline__ float tanhff(float x){ return fmaf(-2.0f, rcpf(1.0f + __expf(2.0f * x)), 1.0f); }

// ---------------- prep: repack weights to f16 [j][256][64], fuse biases -------------
__global__ void rowlstm_prep(const float* __restrict__ i2s_w, const float* __restrict__ i2s_b,
                             const float* __restrict__ s2s_w, const float* __restrict__ s2s_b,
                             h16* __restrict__ wi2s, h16* __restrict__ ws2s,
                             float* __restrict__ biasC) {
  int m = threadIdx.x;  // 0..255 output channel (gate-major)
  for (int j = 0; j < 3; ++j)
    for (int c = 0; c < 64; ++c) {
      wi2s[(j * 256 + m) * 64 + c] = (h16)i2s_w[m * 192 + c * 3 + j];
      ws2s[(j * 256 + m) * 64 + c] = (h16)s2s_w[m * 192 + c * 3 + j];
    }
  biasC[m] = i2s_b[m] + s2s_b[m];
}

// ---------------- conv1: 7x7 VALID, store feat as f16 [b][y][x][c] ------------------
__global__ void rowlstm_conv7(const float* __restrict__ x, const float* __restrict__ w,
                              const float* __restrict__ b, h16* __restrict__ featG) {
  int c  = threadIdx.x;                       // 0..63
  int xo = blockIdx.x * 8 + threadIdx.y;      // 0..127
  int y  = blockIdx.y;
  int bt = blockIdx.z;
  if (xo >= kW) return;
  float a = b[c];
  const float* xp = x + ((size_t)bt * 128 + y) * 128 + xo;
  const float* wp = w + c * 49;
#pragma unroll
  for (int ky = 0; ky < 7; ++ky)
#pragma unroll
    for (int kx = 0; kx < 7; ++kx)
      a = fmaf(xp[ky * 128 + kx], wp[ky * 7 + kx], a);
  featG[(((size_t)bt * kH + y) * kW + xo) * kHid + c] = (h16)a;
}

// ---------------- fused recurrent kernel: one block per batch element ---------------
// 512 threads = 16 waves; wave w owns channels [4w, 4w+4) i.e. gate rows {c,64+c,128+c,192+c}
__global__ __launch_bounds__(512, 1)
void rowlstm_rec(const h16* __restrict__ featG, const h16* __restrict__ wA0,
                 const h16* __restrict__ wA1, const float* __restrict__ biasC,
                 float* __restrict__ out) {
  __shared__ h16 sHT[2][NR * RS];  // h transposed, double buffered (read y-1 / write y)
  __shared__ h16 sFT[2][NR * RS];  // feat row transposed, double buffered (prefetch y+1)

  const int tid  = threadIdx.x;
  const int lane = tid & 31;
  const int wv   = tid >> 5;   // mtile 0..15
  const int nloc = lane & 15;
  const int grp  = lane >> 4;
  const int bb   = blockIdx.x;

  for (int i = tid; i < NR * RS; i += 512) {
    sHT[0][i] = (h16)0.f; sHT[1][i] = (h16)0.f;
    sFT[0][i] = (h16)0.f; sFT[1][i] = (h16)0.f;
  }

  // --- A fragments (resident): a[e<8]=A[m][g*8+e], a[e>=8]=A[m][16+g*8+e-8]
  half16 areg[12];
  {
    const h16* wsrc[2] = {wA0, wA1};
    const int m = nloc;
    const int grow = (m >> 2) * 64 + wv * 4 + (m & 3);  // gate*64 + channel
#pragma unroll
    for (int s = 0; s < 2; ++s)
#pragma unroll
      for (int j = 0; j < 3; ++j)
#pragma unroll
        for (int kt = 0; kt < 2; ++kt) {
          const h16* p = wsrc[s] + (j * 256 + grow) * 64;
          half8 lo = *(const half8*)(p + kt * 32 + grp * 8);
          half8 hi = *(const half8*)(p + kt * 32 + 16 + grp * 8);
          areg[s * 6 + j * 2 + kt] = __builtin_shufflevector(
              lo, hi, 0, 1, 2, 3, 4, 5, 6, 7, 8, 9, 10, 11, 12, 13, 14, 15);
        }
  }

  // --- fused bias per accumulator register (D tile: m = r + 8*grp)
  float bias0[8];
#pragma unroll
  for (int r = 0; r < 8; ++r) {
    int mm = r + 8 * grp;
    bias0[r] = biasC[(mm >> 2) * 64 + wv * 4 + (mm & 3)];
  }

  float cst[8][4];  // c-state: [ntile][channel'] per lane (dup across lane groups)
#pragma unroll
  for (int i = 0; i < 8; ++i)
#pragma unroll
    for (int j = 0; j < 4; ++j) cst[i][j] = 0.f;

  const h16* fbase = featG + (size_t)bb * kH * kW * kHid;

  // stage feat row yy -> sFT[buf] rows 1..122 (976 x 16B chunks, uniform 2 issues/thread)
  auto stage = [&](int yy, int buf) {
    const h16* src = fbase + (size_t)yy * kW * kHid;
#pragma unroll
    for (int it = 0; it < 2; ++it) {
      int idx = tid + it * 512;
      if (idx > 975) idx = 975;  // duplicate copy is benign
      int row = idx >> 3, seg = idx & 7;
      cp16_g2l(&sFT[buf][(row + 1) * RS + seg * 8], src + row * 64 + seg * 8);
    }
  };

  __syncthreads();        // zero-init visible before async writes land
  stage(0, 0);            // prologue: prefetch row 0

  for (int y = 0; y < kH; ++y) {
    const int cur = y & 1;
    __syncthreads();      // all waves done reading sFT[1-cur] / sHT buffers of prev row

    if (y + 1 < kH) {
      stage(y + 1, 1 - cur);   // overlap next row's copy with this row's GEMM
#if HAVE_ASYNC
      __builtin_amdgcn_s_wait_asynccnt(2);  // in-order: oldest batch (row y) complete
#endif
    } else {
#if HAVE_ASYNC
      __builtin_amdgcn_s_wait_asynccnt(0);
#endif
    }
    __syncthreads();      // sFT[cur] + sHT[pr] ready for all waves

    const h16* fb = sFT[cur];
    const h16* hb = sHT[cur];      // h(y-1)
    h16* hw = sHT[1 - cur];        // h(y)

#pragma unroll
    for (int nt = 0; nt < 8; ++nt) {
      // B fragment loader: f = s*6 + j*2 + kt
      auto loadB = [&](int f) -> half16 {
        const int s  = f / 6;
        const int j  = (f % 6) / 2;
        const int kt = f & 1;
        const h16* bbuf = (s == 0) ? fb : hb;
        const h16* bp = bbuf + (nt * 16 + nloc + j) * RS + kt * 32 + grp * 8;
        half8 lo = *(const half8*)bp;
        half8 hi = *(const half8*)(bp + 16);
        return __builtin_shufflevector(lo, hi, 0, 1, 2, 3, 4, 5, 6, 7,
                                       8, 9, 10, 11, 12, 13, 14, 15);
      };

      float8 acc0, acc1;
#pragma unroll
      for (int r = 0; r < 8; ++r) { acc0[r] = bias0[r]; acc1[r] = 0.f; }

      // depth-2 software pipeline over the 12 fragments; two alternating accumulators
      half16 bf[3];
      bf[0] = loadB(0);
      bf[1] = loadB(1);
#pragma unroll
      for (int f = 0; f < 12; ++f) {
        if (f + 2 < 12) bf[(f + 2) % 3] = loadB(f + 2);
        if (f & 1)
          acc1 = __builtin_amdgcn_wmma_f32_16x16x32_f16(
              false, areg[f], false, bf[f % 3], (short)0, acc1, false, false);
        else
          acc0 = __builtin_amdgcn_wmma_f32_16x16x32_f16(
              false, areg[f], false, bf[f % 3], (short)0, acc0, false, false);
      }
      float8 acc = acc0 + acc1;

      // --- LSTM gates: group0 holds {o,f}, group1 holds {i,g}; swap halves once
      float hv[4];
#pragma unroll
      for (int cp = 0; cp < 4; ++cp) {
        float x0 = acc[cp], x1 = acc[cp + 4];
        float z0 = __shfl_xor(x0, 16, 32);
        float z1 = __shfl_xor(x1, 16, 32);
        float og = grp ? z0 : x0;
        float fg = grp ? z1 : x1;
        float ig = grp ? x0 : z0;
        float gg = grp ? x1 : z1;
        float cn = sigf(fg) * cst[nt][cp] + sigf(ig) * tanhff(gg);
        cst[nt][cp] = cn;
        hv[cp] = sigf(og) * tanhff(cn);
      }

      const int ng = nt * 16 + nloc;
      if (grp == 0 && ng < kW) {
        size_t ob = (((size_t)(bb * kHid + wv * 4)) * kH + y) * kW + ng;
        out[ob]                       = hv[0];
        out[ob + (size_t)kH * kW]     = hv[1];
        out[ob + 2 * (size_t)kH * kW] = hv[2];
        out[ob + 3 * (size_t)kH * kW] = hv[3];
        half4 hh = {(h16)hv[0], (h16)hv[1], (h16)hv[2], (h16)hv[3]};
        *(half4*)&hw[(ng + 1) * RS + wv * 4] = hh;  // free transpose into hT
      }
    }
  }
}

extern "C" void kernel_launch(void* const* d_in, const int* in_sizes, int n_in,
                              void* d_out, int out_size, void* d_ws, size_t ws_size,
                              hipStream_t stream) {
  const float* x       = (const float*)d_in[0];
  const float* conv1_w = (const float*)d_in[1];
  const float* conv1_b = (const float*)d_in[2];
  const float* i2s_w   = (const float*)d_in[3];
  const float* i2s_b   = (const float*)d_in[4];
  const float* s2s_w   = (const float*)d_in[5];
  const float* s2s_b   = (const float*)d_in[6];
  float* out = (float*)d_out;

  // workspace layout
  const size_t featBytes = (size_t)kB * kH * kW * kHid * sizeof(h16);  // ~61 MB
  h16*   featG = (h16*)d_ws;
  h16*   wi2s  = (h16*)((char*)d_ws + featBytes);
  h16*   ws2s  = wi2s + 3 * 256 * 64;
  float* biasC = (float*)(ws2s + 3 * 256 * 64);

  rowlstm_prep<<<1, 256, 0, stream>>>(i2s_w, i2s_b, s2s_w, s2s_b, wi2s, ws2s, biasC);
  rowlstm_conv7<<<dim3(16, kH, kB), dim3(64, 8), 0, stream>>>(x, conv1_w, conv1_b, featG);
  rowlstm_rec<<<kB, 512, 0, stream>>>(featG, wi2s, ws2s, biasC, out);
}
